// ODENetwork_39788577030503
// MI455X (gfx1250) — matrix operand order repaired and checked by reading, not compiled, not run on hardware
//
#include <hip/hip_runtime.h>
#include <hip/hip_bf16.h>
#include <stdint.h>

// ---------------------------------------------------------------------------
// Neural ODE (RK2 midpoint, 10 steps) on MI455X / gfx1250.
//   f(z) = tanh(z @ W1 + b1) @ W2 + b2
// 40 sequential ~34-GFLOP GEMMs; all operands L2-resident (192 MB), HBM
// traffic ~80 MB => pure matrix-core problem. bf16 WMMA, fp32 accumulate;
// fp32 RK2 base z kept so rounding does not compound across the 20 evals.
// Pipeline: double-buffered LDS, A staged via global_load_async_to_lds_b128
// (ASYNCcnt), B staged through registers into a WMMA-fragment-major layout
// so every fragment load is a contiguous ds_load_b128.
// ---------------------------------------------------------------------------

typedef __attribute__((ext_vector_type(16))) __bf16 v16bf;
typedef __attribute__((ext_vector_type(8)))  float  v8f;

#define DIM   1024
#define HID   4096
#define BATCH 4096
#define NELEM_Z (BATCH * DIM)
#define NELEM_W (DIM * HID)
#define NSTEPS 10
#define HSTEP  0.1f

#define BM 128
#define BN 128
#define BK 32
#define PITCH 80                 // LDS row pitch bytes (64 data + 16 pad)
#define SH_A  (BM * PITCH)       // 10240 B
#define SH_B  (BN * PITCH)       // 10240 B (B2[col]: 16 packed-pair dwords)

union Frag { v16bf v; uint4 q[2]; };

__device__ __forceinline__ unsigned short f32_to_bf16(float f) {
    union { float f; unsigned u; } x; x.f = f;
    unsigned r = x.u + 0x7FFFu + ((x.u >> 16) & 1u);  // RNE
    return (unsigned short)(r >> 16);
}

__device__ __forceinline__ float fast_tanh(float x) {
#if __has_builtin(__builtin_amdgcn_tanhf)
    return __builtin_amdgcn_tanhf(x);   // V_TANH_F32 (CDNA5 trans op)
#else
    return tanhf(x);
#endif
}

// A tile (128x32 bf16) -> LDS via CDNA5 async copy (tracked by ASYNCcnt).
__device__ __forceinline__ void stageA_async(const unsigned short* A, int K,
                                             int bm, int kc,
                                             unsigned char* shAbuf, int tid) {
    #pragma unroll
    for (int c0 = 0; c0 < 2; ++c0) {
        int c = tid + c0 * 256;            // 512 x 16B chunks
        int row = c >> 2, q = c & 3;
        unsigned long long ga =
            (unsigned long long)(uintptr_t)(A + (size_t)(bm + row) * K + kc + q * 8);
        unsigned ldst = (unsigned)(uintptr_t)(shAbuf + row * PITCH + q * 16);
        asm volatile("global_load_async_to_lds_b128 %0, %1, off"
                     :: "v"(ldst), "v"(ga) : "memory");
    }
}

__device__ __forceinline__ void loadB(const unsigned short* B, int N, int bn,
                                      int kc, int tid, uint4& r0, uint4& r1) {
    int kp = tid & 15;                 // K pair index
    int j8 = (tid >> 4) * 8;           // 8 columns per thread
    r0 = *(const uint4*)(B + (size_t)(kc + 2 * kp)     * N + bn + j8);
    r1 = *(const uint4*)(B + (size_t)(kc + 2 * kp + 1) * N + bn + j8);
}

// B2[col] row = 16 dwords, dword kp = (B[2kp,col], B[2kp+1,col]) packed.
__device__ __forceinline__ void commitB(unsigned char* shBbuf, int tid,
                                        uint4 r0, uint4 r1) {
    int kp = tid & 15;
    int j8 = (tid >> 4) * 8;
    const unsigned short* p0 = (const unsigned short*)&r0;
    const unsigned short* p1 = (const unsigned short*)&r1;
    #pragma unroll
    for (int j = 0; j < 8; ++j) {
        unsigned w = (unsigned)p0[j] | ((unsigned)p1[j] << 16);
        *(unsigned*)(shBbuf + (j8 + j) * PITCH + kp * 4) = w;
    }
}

// ---------------------------------------------------------------------------
//   MODE 0: obf = bf16(tanh(acc + bias[n]))
//   MODE 1: zn = zbase + scale*(acc + bias[n]); obf = bf16(zn); [zout = zn]
// ---------------------------------------------------------------------------
template <int MODE, bool WF32>
__global__ __launch_bounds__(256)
void gemm_wmma(const unsigned short* __restrict__ A,
               const unsigned short* __restrict__ B,
               const float* __restrict__ bias,
               int M, int N, int K, float scale,
               const float* __restrict__ zbase,
               float* __restrict__ zout,
               unsigned short* __restrict__ obf)
{
    __shared__ __align__(16) unsigned char sh[2][SH_A + SH_B];

    const int tid  = threadIdx.x;
    const int lane = tid & 31;
    const int wave = tid >> 5;
    const int wm   = wave & 1;     // 2 wave rows  x 64
    const int wn   = wave >> 1;    // 4 wave cols  x 32
    const int g    = lane >> 4;    // WMMA K-half selector
    const int ln   = lane & 15;

    const int bm = blockIdx.y * BM;
    const int bn = blockIdx.x * BN;

    v8f acc[8];
    #pragma unroll
    for (int i = 0; i < 8; ++i)
        acc[i] = (v8f){0.f, 0.f, 0.f, 0.f, 0.f, 0.f, 0.f, 0.f};

    const int nk = K / BK;
    uint4 br0, br1;
    loadB(B, N, bn, 0, tid, br0, br1);
    stageA_async(A, K, bm, 0, sh[0], tid);

    for (int i = 0; i < nk; ++i) {
        const int cur = i & 1;
        unsigned char* shA = sh[cur];
        unsigned char* shB = sh[cur] + SH_A;

        commitB(shB, tid, br0, br1);                    // waits loadcnt as needed
        asm volatile("s_wait_asynccnt 0" ::: "memory"); // A tile landed in LDS
        __syncthreads();                                // buffer `cur` ready

        if (i + 1 < nk) {                               // prefetch next chunk
            loadB(B, N, bn, (i + 1) * BK, tid, br0, br1);
            stageA_async(A, K, bm, (i + 1) * BK, sh[cur ^ 1], tid);
        }

        // fragments: all contiguous b128 reads (documented wave32 layouts)
        Frag af[4];
        #pragma unroll
        for (int tm = 0; tm < 4; ++tm) {
            const unsigned char* base = shA + (wm * 64 + tm * 16 + ln) * PITCH;
            af[tm].q[0] = *(const uint4*)(base + g * 16);       // K 0-7 / 8-15
            af[tm].q[1] = *(const uint4*)(base + 32 + g * 16);  // K 16-23 / 24-31
        }
        Frag bf2[2];
        #pragma unroll
        for (int tn = 0; tn < 2; ++tn) {
            const unsigned char* bb = shB + (wn * 32 + tn * 16 + ln) * PITCH + g * 16;
            bf2[tn].q[0] = *(const uint4*)(bb);
            bf2[tn].q[1] = *(const uint4*)(bb + 32);
        }

        #pragma unroll
        for (int tm = 0; tm < 4; ++tm)
            #pragma unroll
            for (int tn = 0; tn < 2; ++tn)
                acc[tm * 2 + tn] = __builtin_amdgcn_wmma_f32_16x16x32_bf16(
                    false, af[tm].v, false, bf2[tn].v,
                    (short)0, acc[tm * 2 + tn], false, false);

        __syncthreads();   // all waves done reading `cur` before it is reused
    }

    // branch-free fused epilogue
    #pragma unroll
    for (int tm = 0; tm < 4; ++tm) {
        #pragma unroll
        for (int tn = 0; tn < 2; ++tn) {
            int col  = bn + wn * 32 + tn * 16 + ln;
            float bv = bias[col];
            size_t idx = (size_t)(bm + wm * 64 + tm * 16 + g * 8) * N + col;
            v8f a = acc[tm * 2 + tn];
            #pragma unroll
            for (int r = 0; r < 8; ++r, idx += (size_t)N) {
                float v = a[r] + bv;
                if (MODE == 0) {
                    obf[idx] = f32_to_bf16(fast_tanh(v));
                } else {
                    float zn = zbase[idx] + scale * v;
                    obf[idx] = f32_to_bf16(zn);
                    if (WF32) zout[idx] = zn;
                }
            }
        }
    }
}

__global__ void convert_bf16_kernel(const float* __restrict__ src,
                                    unsigned short* __restrict__ dst, int n) {
    int i = (blockIdx.x * blockDim.x + threadIdx.x) * 4;
    if (i + 3 < n) {
        float4 f = *(const float4*)(src + i);
        unsigned short o[4] = {f32_to_bf16(f.x), f32_to_bf16(f.y),
                               f32_to_bf16(f.z), f32_to_bf16(f.w)};
        *(uint2*)(dst + i) = *(const uint2*)o;
    }
}

__global__ void init_z_kernel(const float* __restrict__ z0,
                              float* __restrict__ zf,
                              unsigned short* __restrict__ zbf,
                              float* __restrict__ out0, int n) {
    int i = (blockIdx.x * blockDim.x + threadIdx.x) * 4;
    if (i + 3 < n) {
        float4 f = *(const float4*)(z0 + i);
        *(float4*)(zf + i)   = f;
        *(float4*)(out0 + i) = f;
        unsigned short o[4] = {f32_to_bf16(f.x), f32_to_bf16(f.y),
                               f32_to_bf16(f.z), f32_to_bf16(f.w)};
        *(uint2*)(zbf + i) = *(const uint2*)o;
    }
}

__global__ void copy_f32_kernel(const float* __restrict__ s,
                                float* __restrict__ d, int n) {
    int i = (blockIdx.x * blockDim.x + threadIdx.x) * 4;
    if (i + 3 < n) *(float4*)(d + i) = *(const float4*)(s + i);
}

extern "C" void kernel_launch(void* const* d_in, const int* in_sizes, int n_in,
                              void* d_out, int out_size, void* d_ws, size_t ws_size,
                              hipStream_t stream) {
    const float* z0 = (const float*)d_in[0];
    // d_in[1] = t = [0,1] -> 10 RK2 steps of h = 0.1 (fixed by problem setup)
    const float* W1 = (const float*)d_in[2];
    const float* b1 = (const float*)d_in[3];
    const float* W2 = (const float*)d_in[4];
    const float* b2 = (const float*)d_in[5];
    float* out = (float*)d_out;

    char* ws = (char*)d_ws;
    unsigned short* W1bf = (unsigned short*)(ws);                                //  8 MB
    unsigned short* W2bf = (unsigned short*)(ws + (size_t)8  * 1024 * 1024);     //  8 MB
    float*          zf   = (float*)        (ws + (size_t)16 * 1024 * 1024);      // 16 MB
    unsigned short* zbf  = (unsigned short*)(ws + (size_t)32 * 1024 * 1024);     //  8 MB
    unsigned short* Hbf  = (unsigned short*)(ws + (size_t)40 * 1024 * 1024);     // 32 MB

    const int cthreads = 256;
    convert_bf16_kernel<<<NELEM_W / 4 / cthreads, cthreads, 0, stream>>>(W1, W1bf, NELEM_W);
    convert_bf16_kernel<<<NELEM_W / 4 / cthreads, cthreads, 0, stream>>>(W2, W2bf, NELEM_W);
    init_z_kernel<<<NELEM_Z / 4 / cthreads, cthreads, 0, stream>>>(z0, zf, zbf, out, NELEM_Z);

    dim3 grid1(HID / BN, BATCH / BM);   // 32 x 32  (z @ W1)
    dim3 grid2(DIM / BN, BATCH / BM);   //  8 x 32  (h @ W2)

    for (int s = 0; s < NSTEPS; ++s) {
        // k1: H = tanh(z@W1+b1); z_mid(bf16) = z + (h/2)*(H@W2+b2)
        gemm_wmma<0, false><<<grid1, 256, 0, stream>>>(zbf, W1bf, b1, BATCH, HID, DIM,
                                                       0.f, nullptr, nullptr, Hbf);
        gemm_wmma<1, false><<<grid2, 256, 0, stream>>>(Hbf, W2bf, b2, BATCH, DIM, HID,
                                                       HSTEP * 0.5f, zf, nullptr, zbf);
        // k2: H = tanh(z_mid@W1+b1); z = z + h*(H@W2+b2)  (fp32 + bf16)
        gemm_wmma<0, false><<<grid1, 256, 0, stream>>>(zbf, W1bf, b1, BATCH, HID, DIM,
                                                       0.f, nullptr, nullptr, Hbf);
        gemm_wmma<1, true><<<grid2, 256, 0, stream>>>(Hbf, W2bf, b2, BATCH, DIM, HID,
                                                      HSTEP, zf, zf, zbf);
    }

    copy_f32_kernel<<<NELEM_Z / 4 / cthreads, cthreads, 0, stream>>>(
        zf, out + (size_t)NELEM_Z, NELEM_Z);
}
